// GATConvTransformerDecoderLayer_44641890075097
// MI455X (gfx1250) — matrix operand order repaired and checked by reading, not compiled, not run on hardware
//
#include <hip/hip_runtime.h>
#include <math.h>

// ---------------------------------------------------------------------------
// CDNA5 (gfx1250) wave32 WMMA types
// ---------------------------------------------------------------------------
typedef __attribute__((ext_vector_type(16))) __bf16          v16bf;
typedef __attribute__((ext_vector_type(8)))  float           v8f;
typedef __attribute__((ext_vector_type(8)))  unsigned short  u16x8;

union BF16Frag {
    v16bf          v;
    u16x8          h8[2];
    unsigned short u[16];
};

// Native bf16 convert (v_cvt_pk_bf16_f32 on gfx1250)
__device__ __forceinline__ unsigned short bfbits(float f) {
    __bf16 b = (__bf16)f;
    return __builtin_bit_cast(unsigned short, b);
}

// convert two float4 into 8 consecutive v16bf elements
__device__ __forceinline__ void cvt8(v16bf& f, int base, float4 a, float4 b) {
    f[base + 0] = (__bf16)a.x; f[base + 1] = (__bf16)a.y;
    f[base + 2] = (__bf16)a.z; f[base + 3] = (__bf16)a.w;
    f[base + 4] = (__bf16)b.x; f[base + 5] = (__bf16)b.y;
    f[base + 6] = (__bf16)b.z; f[base + 7] = (__bf16)b.w;
}

__device__ __forceinline__ void atomicMaxFloat(float* addr, float v) {
    // classic ordered-bits trick; works with the -1e30f init value
    if (v >= 0.0f) atomicMax((int*)addr, __float_as_int(v));
    else           atomicMin((unsigned int*)addr, __float_as_uint(v));
}

// LDS byte offset of a __shared__ object (flat-pointer low 32 bits == LDS addr)
__device__ __forceinline__ unsigned lds_off(const void* p) {
    return (unsigned)(unsigned long long)p;
}

// ---------------------------------------------------------------------------
// bf16 WMMA GEMM:  C[M,Ncols] = act(A[M,K] * B^T + bias)
//   B is [Ncols, ldb] row-major (ldb == K) -> every fragment load contiguous.
//   outBf16=1 stores bf16 (for QKV feeding flash-attention), else f32.
// One wave computes a 16x64 tile. All raw loads for a k-step are issued
// before any conversion so the loads clause and overlap (MLP), then the
// packed bf16 converts feed 4 back-to-back WMMAs.
// ---------------------------------------------------------------------------
__global__ __launch_bounds__(128) void gemm_wmma_bf16(
    const float* __restrict__ A, const float* __restrict__ B,
    const float* __restrict__ bias, void* __restrict__ Cout,
    int M, int Ncols, int K, int ldb, int act, int outBf16)
{
    const int lane = threadIdx.x & 31;
    const int wv   = threadIdx.x >> 5;
    const int l15  = lane & 15;
    const int hh   = lane >> 4;            // half-wave select
    const int tilesN = Ncols >> 6;
    const int widx = blockIdx.x * 4 + wv;
    const int tm   = (widx / tilesN) << 4;
    const int tn   = (widx % tilesN) << 6;
    if (tm >= M) return;

    v8f zero = {};
    v8f acc[4];
#pragma unroll
    for (int t = 0; t < 4; ++t) acc[t] = zero;

    const float* Arow = A + (size_t)(tm + l15) * K;

    for (int k0 = 0; k0 < K; k0 += 32) {
        if (k0 + 32 < K) __builtin_prefetch(Arow + k0 + 32, 0, 0);

        // ---- load phase: issue all 20 b128 loads before any consumption ----
        // A fragment 16x32: elems 0..7 -> k = 8*hh+i ; 8..15 -> k = 16+8*hh+i
        const float* Ap = Arow + k0;
        float4 ar[4];
        ar[0] = *(const float4*)(Ap + 8 * hh);
        ar[1] = *(const float4*)(Ap + 8 * hh + 4);
        ar[2] = *(const float4*)(Ap + 16 + 8 * hh);
        ar[3] = *(const float4*)(Ap + 16 + 8 * hh + 4);
        // B fragments 32x16: elems 0..7 -> k = 8*hh+i ; 8..15 -> k = 16+8*hh+i
        float4 br[4][4];
#pragma unroll
        for (int t = 0; t < 4; ++t) {
            const float* Bp = B + (size_t)(tn + t * 16 + l15) * ldb + k0 + 8 * hh;
#pragma unroll
            for (int c = 0; c < 4; ++c) br[t][c] = *(const float4*)(Bp + 4 * c + 8 * (c >> 1));
        }

        // ---- convert + WMMA phase ----
        v16bf fa;
        cvt8(fa, 0, ar[0], ar[1]);
        cvt8(fa, 8, ar[2], ar[3]);
#pragma unroll
        for (int t = 0; t < 4; ++t) {
            v16bf fb;
            cvt8(fb, 0, br[t][0], br[t][1]);
            cvt8(fb, 8, br[t][2], br[t][3]);
            acc[t] = __builtin_amdgcn_wmma_f32_16x16x32_bf16(
                false, fa, false, fb, (short)0, acc[t], false, false);
        }
    }

#pragma unroll
    for (int t = 0; t < 4; ++t) {
        const int col = tn + t * 16 + l15;
        const float bv = bias ? bias[col] : 0.0f;
#pragma unroll
        for (int v = 0; v < 8; ++v) {
            const int row = tm + v + 8 * hh;     // C layout: row = v + 8*half
            float x = acc[t][v] + bv;
            if (act == 1) x = x > 0.0f ? x : 0.0f;
            if (outBf16)
                ((unsigned short*)Cout)[(size_t)row * Ncols + col] = bfbits(x);
            else
                ((float*)Cout)[(size_t)row * Ncols + col] = x;
        }
    }
}

// Transpose W_gat [128,256] -> [256,128] so the GEMM only needs the B^T path.
__global__ void transpose_128x256(const float* __restrict__ in, float* __restrict__ out)
{
    int i = blockIdx.x * blockDim.x + threadIdx.x;   // 32768 elements
    if (i >= 128 * 256) return;
    const int r = i >> 8, c = i & 255;
    out[c * 128 + r] = in[i];
}

// ---------------------------------------------------------------------------
// GAT helper kernels
// ---------------------------------------------------------------------------
__global__ void gat_as_ad(const float* __restrict__ xh,
                          const float* __restrict__ att_src,
                          const float* __restrict__ att_dst,
                          float* __restrict__ a_s, float* __restrict__ a_d, int NH)
{
    int i = blockIdx.x * blockDim.x + threadIdx.x;
    if (i >= NH) return;
    const int n = i >> 2, h = i & 3;
    const float* xp = xh + (size_t)n * 256 + h * 64;
    const float* as = att_src + h * 64;
    const float* ad = att_dst + h * 64;
    float s = 0.f, d = 0.f;
#pragma unroll 8
    for (int c = 0; c < 64; ++c) { s += xp[c] * as[c]; d += xp[c] * ad[c]; }
    a_s[i] = s; a_d[i] = d;
}

__global__ void gat_init(float* __restrict__ mbuf, float* __restrict__ dbuf,
                         float* __restrict__ acc, int NH, int NE)
{
    int i = blockIdx.x * blockDim.x + threadIdx.x;
    if (i < NH) { mbuf[i] = -1e30f; dbuf[i] = 0.0f; }
    if (i < NE) acc[i] = 0.0f;
}

__device__ __forceinline__ void edge_decode(const int* ei, int E, int e, int& s, int& d) {
    if (e < E) { s = ei[e]; d = ei[E + e]; }
    else       { s = d = e - E; }              // self loop
}

__global__ void gat_edge_max(const int* __restrict__ ei, int E, int ET,
                             const float* __restrict__ a_s, const float* __restrict__ a_d,
                             float* __restrict__ mbuf)
{
    int e = blockIdx.x * blockDim.x + threadIdx.x;
    if (e >= ET) return;
    int s, d; edge_decode(ei, E, e, s, d);
#pragma unroll
    for (int h = 0; h < 4; ++h) {
        float a = a_s[s * 4 + h] + a_d[d * 4 + h];
        a = a > 0.0f ? a : 0.2f * a;           // leaky_relu(0.2)
        atomicMaxFloat(&mbuf[d * 4 + h], a);
    }
}

__global__ void gat_edge_exp(const int* __restrict__ ei, int E, int ET,
                             const float* __restrict__ a_s, const float* __restrict__ a_d,
                             const float* __restrict__ mbuf,
                             float* __restrict__ pbuf, float* __restrict__ dbuf)
{
    int e = blockIdx.x * blockDim.x + threadIdx.x;
    if (e >= ET) return;
    int s, d; edge_decode(ei, E, e, s, d);
#pragma unroll
    for (int h = 0; h < 4; ++h) {
        float a = a_s[s * 4 + h] + a_d[d * 4 + h];
        a = a > 0.0f ? a : 0.2f * a;
        float p = __expf(a - mbuf[d * 4 + h]);
        pbuf[e * 4 + h] = p;
        atomicAdd(&dbuf[d * 4 + h], p);
    }
}

__global__ void gat_scatter(const int* __restrict__ ei, int E, int ET,
                            const float* __restrict__ xh,
                            const float* __restrict__ pbuf, const float* __restrict__ dbuf,
                            float* __restrict__ acc)
{
    int i = blockIdx.x * blockDim.x + threadIdx.x;
    if (i >= ET * 4) return;
    const int e = i >> 2, h = i & 3;
    int s, d; edge_decode(ei, E, e, s, d);
    const float w = pbuf[i] / (dbuf[d * 4 + h] + 1e-16f);
    const float* xp = xh + (size_t)s * 256 + h * 64;
    float* op = acc + (size_t)d * 256 + h * 64;
#pragma unroll 4
    for (int c = 0; c < 64; ++c) atomicAdd(&op[c], w * xp[c]);
}

__global__ void gat_finalize(const float* __restrict__ acc, const float* __restrict__ bias,
                             float* __restrict__ tbuf, int total)
{
    int i = blockIdx.x * blockDim.x + threadIdx.x;
    if (i >= total) return;
    const float x = acc[i] + bias[i & 255];
    tbuf[i] = x > 0.0f ? x : 0.01f * x;        // F.leaky_relu default
}

// ---------------------------------------------------------------------------
// Flash attention (bf16 WMMA, f32 accum). Block = 4 waves; each wave owns a
// 16-row Q tile of one head. K tile and raw V tile are staged into LDS with
// gfx1250 async-to-LDS loads, double-buffered on the ASYNCcnt pipe so the
// next tile's global traffic overlaps the current tile's 8 WMMAs + softmax.
// ---------------------------------------------------------------------------
__global__ __launch_bounds__(128) void flash_attn_kernel(
    const unsigned short* __restrict__ qbf,
    const unsigned short* __restrict__ kbf,
    const unsigned short* __restrict__ vbf,
    float* __restrict__ obuf, int N)
{
    __shared__ __attribute__((aligned(32))) unsigned short kT[2][32 * 64];   // [key][d]
    __shared__ __attribute__((aligned(32))) unsigned short vR[2][32 * 64];   // [key][d] raw
    __shared__ __attribute__((aligned(32))) unsigned short vT[2][64 * 32];   // [d][key]
    __shared__ __attribute__((aligned(32))) unsigned short pS[4][16 * 32];   // per-wave [row][key]

    const int h    = blockIdx.y;
    const int lane = threadIdx.x & 31;
    const int wv   = threadIdx.x >> 5;
    const int l15  = lane & 15;
    const int hh   = lane >> 4;
    const int qrow0 = blockIdx.x * 64 + wv * 16;

    // Q fragments: HD=64 split into two K=32 A-fragments, loaded once.
    BF16Frag aq0, aq1;
    {
        const unsigned short* qp = qbf + (size_t)(qrow0 + l15) * 256 + h * 64;
        aq0.h8[0] = *(const u16x8*)(qp + 16 * hh);
        aq0.h8[1] = *(const u16x8*)(qp + 16 * hh + 8);
        aq1.h8[0] = *(const u16x8*)(qp + 32 + 16 * hh);
        aq1.h8[1] = *(const u16x8*)(qp + 32 + 16 * hh + 8);
    }

    float mrow[8], lrow[8];
    v8f zero = {};
    v8f oa[4];
#pragma unroll
    for (int v = 0; v < 8; ++v) { mrow[v] = -1e30f; lrow[v] = 0.0f; }
#pragma unroll
    for (int t = 0; t < 4; ++t) oa[t] = zero;

    const int key_c = threadIdx.x >> 2;        // 0..31  (cooperative loader)
    const int dch   = (threadIdx.x & 3) * 16;  // 0,16,32,48

    // per-thread LDS destinations for async staging (32B each, per buffer)
    const unsigned ldsK[2] = { lds_off(&kT[0][key_c * 64 + dch]),
                               lds_off(&kT[1][key_c * 64 + dch]) };
    const unsigned ldsV[2] = { lds_off(&vR[0][key_c * 64 + dch]),
                               lds_off(&vR[1][key_c * 64 + dch]) };

    // GVS async-to-LDS: mem = SGPR base + VGPR offset + inst_offset;
    // inst_offset also applies to the LDS destination (ISA ch.10 async pseudocode)
    auto issue_tile = [&](int j, int b) {
        const unsigned goff =
            ((unsigned)(j + key_c) * 256u + (unsigned)h * 64u + (unsigned)dch) * 2u;
        asm volatile("global_load_async_to_lds_b128 %0, %1, %2 offset:0"
                     :: "v"(ldsK[b]), "v"(goff), "s"(kbf) : "memory");
        asm volatile("global_load_async_to_lds_b128 %0, %1, %2 offset:16"
                     :: "v"(ldsK[b]), "v"(goff), "s"(kbf) : "memory");
        asm volatile("global_load_async_to_lds_b128 %0, %1, %2 offset:0"
                     :: "v"(ldsV[b]), "v"(goff), "s"(vbf) : "memory");
        asm volatile("global_load_async_to_lds_b128 %0, %1, %2 offset:16"
                     :: "v"(ldsV[b]), "v"(goff), "s"(vbf) : "memory");
    };

    issue_tile(0, 0);                          // prologue: 4 async ops in flight

    for (int j = 0; j < N; j += 32) {
        const int cur = (j >> 5) & 1;
        __syncthreads();                       // buffer cur^1 free for reuse
        if (j + 32 < N) {
            issue_tile(j + 32, cur ^ 1);       // 8 in flight
            // async loads complete in order: <=4 outstanding => cur tile landed
            asm volatile("s_wait_asynccnt 0x4" ::: "memory");
        } else {
            asm volatile("s_wait_asynccnt 0x0" ::: "memory");
        }
        {   // transpose the 16 V values this thread's own async brought in
            const unsigned short* vr = &vR[cur][key_c * 64 + dch];
            u16x8 va = *(const u16x8*)vr;
            u16x8 vb = *(const u16x8*)(vr + 8);
#pragma unroll
            for (int i = 0; i < 8; ++i) {
                vT[cur][(dch + i) * 32 + key_c]     = va[i];
                vT[cur][(dch + 8 + i) * 32 + key_c] = vb[i];
            }
        }
        __syncthreads();                       // publish kT[cur], vT[cur]

        // S = Q K^T (scale applied in softmax below) for two 16-key sub-tiles
        v8f s[2];
#pragma unroll
        for (int t = 0; t < 2; ++t) {
            const unsigned short* kr = &kT[cur][(t * 16 + l15) * 64];
            v16bf b0 = *(const v16bf*)(kr + 16 * hh);        // d = 0..31
            v16bf b1 = *(const v16bf*)(kr + 32 + 16 * hh);   // d = 32..63
            v8f c = zero;
            c = __builtin_amdgcn_wmma_f32_16x16x32_bf16(false, aq0.v, false, b0, (short)0, c, false, false);
            c = __builtin_amdgcn_wmma_f32_16x16x32_bf16(false, aq1.v, false, b1, (short)0, c, false, false);
            s[t] = c;
        }

        // online softmax (rows live in C layout: row = v + 8*hh, col = l15)
        float p0[8], p1[8], corr[8];
#pragma unroll
        for (int v = 0; v < 8; ++v) {
            const float s0 = s[0][v] * 0.125f;   // 1/sqrt(64)
            const float s1 = s[1][v] * 0.125f;
            float r = fmaxf(s0, s1);
            r = fmaxf(r, __shfl_xor(r, 1, 32));
            r = fmaxf(r, __shfl_xor(r, 2, 32));
            r = fmaxf(r, __shfl_xor(r, 4, 32));
            r = fmaxf(r, __shfl_xor(r, 8, 32));
            const float mn = fmaxf(mrow[v], r);
            const float cc = __expf(mrow[v] - mn);
            const float e0 = __expf(s0 - mn);
            const float e1 = __expf(s1 - mn);
            float rs = e0 + e1;
            rs += __shfl_xor(rs, 1, 32);
            rs += __shfl_xor(rs, 2, 32);
            rs += __shfl_xor(rs, 4, 32);
            rs += __shfl_xor(rs, 8, 32);
            lrow[v] = lrow[v] * cc + rs;
            mrow[v] = mn;
            corr[v] = cc; p0[v] = e0; p1[v] = e1;
        }
#pragma unroll
        for (int t = 0; t < 4; ++t)
#pragma unroll
            for (int v = 0; v < 8; ++v) oa[t][v] *= corr[v];

        // transpose P (C layout) -> A layout through per-wave LDS scratch
        unsigned short* pw = pS[wv];
#pragma unroll
        for (int v = 0; v < 8; ++v) {
            const int row = v + 8 * hh;
            pw[row * 32 + l15]      = bfbits(p0[v]);
            pw[row * 32 + 16 + l15] = bfbits(p1[v]);
        }
        __syncthreads();   // order cross-lane LDS writes before reads (uniform)

        BF16Frag ap;
        {
            const unsigned short* pr = pw + l15 * 32;
            ap.h8[0] = *(const u16x8*)(pr + 16 * hh);
            ap.h8[1] = *(const u16x8*)(pr + 16 * hh + 8);
        }
        // O += P V  : B frags from transposed V tile, contiguous LDS reads
#pragma unroll
        for (int t = 0; t < 4; ++t) {
            v16bf bv = *(const v16bf*)&vT[cur][(t * 16 + l15) * 32 + 16 * hh];
            oa[t] = __builtin_amdgcn_wmma_f32_16x16x32_bf16(false, ap.v, false, bv, (short)0, oa[t], false, false);
        }
    }

    // normalize and store
#pragma unroll
    for (int t = 0; t < 4; ++t)
#pragma unroll
        for (int v = 0; v < 8; ++v) {
            const int row = qrow0 + v + 8 * hh;
            obuf[(size_t)row * 256 + h * 64 + t * 16 + l15] = oa[t][v] / lrow[v];
        }
}

// ---------------------------------------------------------------------------
// Orchestration
// ---------------------------------------------------------------------------
extern "C" void kernel_launch(void* const* d_in, const int* in_sizes, int n_in,
                              void* d_out, int out_size, void* d_ws, size_t ws_size,
                              hipStream_t stream)
{
    const float* tgt        = (const float*)d_in[0];
    const float* memory     = (const float*)d_in[1];
    const float* orig       = (const float*)d_in[2];
    const int*   ei         = (const int*)  d_in[3];
    const float* W_gat      = (const float*)d_in[4];
    const float* att_src    = (const float*)d_in[5];
    const float* att_dst    = (const float*)d_in[6];
    const float* b_gat      = (const float*)d_in[7];
    const float* in_proj_w  = (const float*)d_in[8];
    const float* in_proj_b  = (const float*)d_in[9];
    const float* out_proj_w = (const float*)d_in[10];
    const float* out_proj_b = (const float*)d_in[11];
    const float* lin1_w     = (const float*)d_in[12];
    const float* lin1_b     = (const float*)d_in[13];

    const int N  = in_sizes[0] / 128;   // 4096
    const int E  = in_sizes[3] / 2;     // 262144
    const int ET = E + N;               // edges + self loops

    char* w = (char*)d_ws;
    auto carve = [&](size_t bytes) {
        char* p = w; w += (bytes + 255) & ~(size_t)255; return p;
    };
    float* xh  = (float*)carve((size_t)N * 256 * 4);   // GAT lin output [N,H,C]
    float* tb  = (float*)carve((size_t)N * 256 * 4);   // GAT output t
    float* acc = (float*)carve((size_t)N * 256 * 4);   // scatter accumulator
    float* ob  = (float*)carve((size_t)N * 256 * 4);   // attention output
    float* fu  = (float*)carve((size_t)N * 256 * 4);   // out-proj output
    float* a_s = (float*)carve((size_t)N * 4 * 4);
    float* a_d = (float*)carve((size_t)N * 4 * 4);
    float* mb  = (float*)carve((size_t)N * 4 * 4);
    float* db  = (float*)carve((size_t)N * 4 * 4);
    float* pb  = (float*)carve((size_t)ET * 4 * 4);
    float* wgT = (float*)carve((size_t)256 * 128 * 4); // W_gat transposed
    unsigned short* qbf = (unsigned short*)carve((size_t)N * 256 * 2);
    unsigned short* kbf = (unsigned short*)carve((size_t)N * 256 * 2);
    unsigned short* vbf = (unsigned short*)carve((size_t)N * 256 * 2);

    const int thr = 256;
    // 0) transpose W_gat so all GEMMs use the contiguous B^T path
    transpose_128x256<<<(128 * 256 + thr - 1) / thr, thr, 0, stream>>>(W_gat, wgT);
    // 1) xh = tgt @ W_gat         (M=N, Ncols=256, K=128)
    gemm_wmma_bf16<<<(N / 16) * (256 / 64) / 4, 128, 0, stream>>>(
        tgt, wgT, nullptr, xh, N, 256, 128, 128, 0, 0);
    // 2) attention logits per node/head
    gat_as_ad<<<(N * 4 + thr - 1) / thr, thr, 0, stream>>>(xh, att_src, att_dst, a_s, a_d, N * 4);
    gat_init<<<(N * 256 + thr - 1) / thr, thr, 0, stream>>>(mb, db, acc, N * 4, N * 256);
    // 3) edge softmax (max / exp+sum / weighted scatter)
    gat_edge_max<<<(ET + thr - 1) / thr, thr, 0, stream>>>(ei, E, ET, a_s, a_d, mb);
    gat_edge_exp<<<(ET + thr - 1) / thr, thr, 0, stream>>>(ei, E, ET, a_s, a_d, mb, pb, db);
    gat_scatter<<<(ET * 4 + thr - 1) / thr, thr, 0, stream>>>(ei, E, ET, xh, pb, db, acc);
    gat_finalize<<<(N * 256 + thr - 1) / thr, thr, 0, stream>>>(acc, b_gat, tb, N * 256);

    // 4) QKV projections (write bf16 directly for the attention kernel)
    const int g256 = (N / 16) * (256 / 64) / 4;
    gemm_wmma_bf16<<<g256, 128, 0, stream>>>(tb,     in_proj_w,             in_proj_b,       qbf, N, 256, 256, 256, 0, 1);
    gemm_wmma_bf16<<<g256, 128, 0, stream>>>(memory, in_proj_w + 256 * 256, in_proj_b + 256, kbf, N, 256, 256, 256, 0, 1);
    gemm_wmma_bf16<<<g256, 128, 0, stream>>>(orig,   in_proj_w + 512 * 256, in_proj_b + 512, vbf, N, 256, 256, 256, 0, 1);

    // 5) flash attention: grid = (qblocks, heads)
    dim3 ag(N / 64, 4);
    flash_attn_kernel<<<ag, 128, 0, stream>>>(qbf, kbf, vbf, ob, N);

    // 6) output projection, then FFN linear1 + ReLU -> d_out
    gemm_wmma_bf16<<<g256, 128, 0, stream>>>(ob, out_proj_w, out_proj_b, fu, N, 256, 256, 256, 0, 0);
    gemm_wmma_bf16<<<(N / 16) * (128 / 64) / 4, 128, 0, stream>>>(
        fu, lin1_w, lin1_b, d_out, N, 128, 256, 256, 1, 0);
}
